// AVWGCN_19250043421414
// MI455X (gfx1250) — compile-verified, matrix operand order checked by reading, not run on hardware
//
#include <hip/hip_runtime.h>
#include <hip/hip_bf16.h>

// ---------------------------------------------------------------------------
// AVWGCN on MI455X (gfx1250): bf16 WMMA pipeline.
//   S = softmax(relu(E E^T))                      [2048 x 2048]
//   Y1 = S @ X_bt        (per bt in 0..191)       [bf16 WMMA, f32 acc]
//   Y2 = 2 S @ Y1 - X
//   out[bt,n,:] = sum_{k,i} G[n,k,bt,i] * W[n,k,i,:] + bias[n,:]
// Workspace layout (needs ~260 MB):
//   Sb  : [2048][2048] bf16                 (A operand, row/K-contiguous)
//   Xt  : [192][64][2048] bf16              (B operand for S@X, K=m contiguous)
//   Y1t : [192][64][2048] bf16              (B operand for S@Y1)
//   G   : [2048][3][192][64] bf16           (A operand for grouped GEMM, K=i contig)
// ---------------------------------------------------------------------------

#define NNODE 2048
#define EMB   16
#define DD    64
#define TT    12
#define BB    16
#define BT    (BB*TT)          // 192
#define KCH   3
#define XT_BT (DD*NNODE)       // 131072 elements per bt slice of Xt/Y1t
#define G_N   (KCH*BT*DD)      // 36864 elements per node in G
#define G_K   (BT*DD)          // 12288 elements per k-hop in G
#define TOTX  (BB*NNODE*TT*DD) // 25165824

typedef __attribute__((ext_vector_type(16))) __bf16       v16bf;
typedef __attribute__((ext_vector_type(8)))  float        v8f;
typedef __attribute__((ext_vector_type(4)))  unsigned int u32x4;

// One WMMA fragment (A or B, 16x32 bf16) = two contiguous 16B runs per lane.
__device__ __forceinline__ v16bf load_frag(const __bf16* p) {
    union { u32x4 u[2]; v16bf v; } f;
    f.u[0] = *reinterpret_cast<const u32x4*>(p);
    f.u[1] = *reinterpret_cast<const u32x4*>(p + 16);  // halves 8..15 -> K+16..K+23
    return f.v;
}

// ---------------------------------------------------------------------------
// K1: pack x (f32 [B][N][T][D]) into Xt[bt][c][m] bf16 and G[n][0][bt][c] bf16
// ---------------------------------------------------------------------------
__global__ void pack_x_kernel(const float* __restrict__ x,
                              __bf16* __restrict__ Xt,
                              __bf16* __restrict__ G) {
    int i = blockIdx.x * blockDim.x + threadIdx.x;
    if (i >= TOTX) return;
    int d = i & (DD - 1);
    int t = (i >> 6) % TT;
    int n = (i / (DD * TT)) & (NNODE - 1);
    int b = i / (DD * TT * NNODE);
    float v = x[i];
    int bt = b * TT + t;
    Xt[bt * XT_BT + d * NNODE + n] = (__bf16)v;   // K(m)-contiguous for B operand
    G[n * G_N + bt * DD + d]       = (__bf16)v;   // k=0 hop (identity support)
}

// ---------------------------------------------------------------------------
// K2: S row softmax(relu(E E^T)) -> bf16.  One block per row, 256 threads x 8.
// ---------------------------------------------------------------------------
__global__ void supports_kernel(const float* __restrict__ E,
                                __bf16* __restrict__ S) {
    const int n = blockIdx.x;
    const int tid = threadIdx.x;
    __shared__ float En[EMB];
    __shared__ float red[256];
    if (tid < EMB) En[tid] = E[n * EMB + tid];
    __syncthreads();

    float v[8];
    float mx = 0.0f;                      // relu => 0 is a valid max floor
    #pragma unroll
    for (int j = 0; j < 8; ++j) {
        int m = tid * 8 + j;
        const float* Em = E + m * EMB;
        float d = 0.0f;
        #pragma unroll
        for (int e = 0; e < EMB; ++e) d += En[e] * Em[e];
        d = fmaxf(d, 0.0f);
        v[j] = d;
        mx = fmaxf(mx, d);
    }
    red[tid] = mx; __syncthreads();
    for (int s = 128; s > 0; s >>= 1) {
        if (tid < s) red[tid] = fmaxf(red[tid], red[tid + s]);
        __syncthreads();
    }
    mx = red[0]; __syncthreads();

    float sum = 0.0f;
    #pragma unroll
    for (int j = 0; j < 8; ++j) { v[j] = __expf(v[j] - mx); sum += v[j]; }
    red[tid] = sum; __syncthreads();
    for (int s = 128; s > 0; s >>= 1) {
        if (tid < s) red[tid] += red[tid + s];
        __syncthreads();
    }
    float inv = 1.0f / red[0];
    #pragma unroll
    for (int j = 0; j < 8; ++j)
        S[n * NNODE + tid * 8 + j] = (__bf16)(v[j] * inv);
}

// ---------------------------------------------------------------------------
// K3/K4: Y = S @ B  (B-src K-contiguous).  grid = (32 m-tiles, 192 bt).
// 128 threads = 4 waves, each wave owns a 16x64 output tile; K loop 2048/32.
// MODE 1: write Y1t (B layout) and G[n][1].  MODE 2: Y2 = 2*acc - x -> G[n][2].
// ---------------------------------------------------------------------------
template <int MODE>
__global__ void __launch_bounds__(128)
gemm_s_kernel(const __bf16* __restrict__ S,
              const __bf16* __restrict__ Bsrc,
              const float*  __restrict__ xorig,
              __bf16* __restrict__ OutT,
              __bf16* __restrict__ G) {
    const int mtile = blockIdx.x;             // 0..31
    const int bt    = blockIdx.y;             // 0..191
    const int wave  = threadIdx.x >> 5;
    const int lane  = threadIdx.x & 31;
    const int lm    = lane & 15;
    const int lk    = (lane >> 4) << 3;       // 0 or 8
    const int m0    = mtile * 64 + wave * 16;

    const __bf16* Arow  = S + (m0 + lm) * NNODE + lk;
    const __bf16* Bbase = Bsrc + bt * XT_BT + lm * NNODE + lk;

    v8f acc[4] = {};
    for (int k0 = 0; k0 < NNODE; k0 += 32) {
        if (k0 + 256 < NNODE) {
            __builtin_prefetch(Arow + k0 + 256, 0, 1);
            __builtin_prefetch(Bbase + k0 + 256, 0, 1);
        }
        v16bf a = load_frag(Arow + k0);
        #pragma unroll
        for (int f = 0; f < 4; ++f) {
            v16bf b = load_frag(Bbase + f * 16 * NNODE + k0);
            acc[f] = __builtin_amdgcn_wmma_f32_16x16x32_bf16(
                false, a, false, b, (short)0, acc[f], false, false);
        }
    }

    const int mrow8 = (lane >> 4) << 3;       // D-matrix row offset for hi lanes
    if (MODE == 1) {
        #pragma unroll
        for (int f = 0; f < 4; ++f) {
            int c = f * 16 + lm;
            #pragma unroll
            for (int r = 0; r < 8; ++r) {
                int M = m0 + r + mrow8;
                __bf16 h = (__bf16)acc[f][r];
                OutT[bt * XT_BT + c * NNODE + M] = h;      // B operand for hop 2
                G[M * G_N + G_K + bt * DD + c]   = h;      // k=1 hop
            }
        }
    } else {
        const int b = bt / TT, t = bt % TT;
        #pragma unroll
        for (int f = 0; f < 4; ++f) {
            int c = f * 16 + lm;
            #pragma unroll
            for (int r = 0; r < 8; ++r) {
                int M = m0 + r + mrow8;
                float xv = xorig[(((b * NNODE + M) * TT) + t) * DD + c];
                G[M * G_N + 2 * G_K + bt * DD + c] = (__bf16)(2.0f * acc[f][r] - xv);
            }
        }
    }
}

// ---------------------------------------------------------------------------
// K5: per-node hypernet + grouped GEMM.  One block per node, 128 threads.
//   Wlds[kk][o][i] bf16 (i-contiguous -> B frags via ds_load_b128)
//   A frags from G[n][kk][bt][i] (i-contiguous).
// ---------------------------------------------------------------------------
__global__ void __launch_bounds__(128)
final_kernel(const float* __restrict__ E,
             const float* __restrict__ Wp,
             const float* __restrict__ biasP,
             const __bf16* __restrict__ G,
             float* __restrict__ out) {
    const int n   = blockIdx.x;
    const int tid = threadIdx.x;
    __shared__ __align__(16) __bf16 Wlds[KCH * DD * DD];  // 24.6 KB
    __shared__ float biasS[DD];
    __shared__ float EnS[EMB];

    if (tid < EMB) EnS[tid] = E[n * EMB + tid];
    __syncthreads();

    for (int idx = tid; idx < KCH * DD * DD; idx += 128) {
        int kk  = idx / (DD * DD);
        int rem = idx - kk * DD * DD;
        int i   = rem >> 6;
        int o   = rem & (DD - 1);
        float a = 0.0f;
        #pragma unroll
        for (int e = 0; e < EMB; ++e)
            a += EnS[e] * Wp[((e * KCH + kk) * DD + i) * DD + o];
        Wlds[(kk * DD + o) * DD + i] = (__bf16)a;         // i-contiguous
    }
    if (tid < DD) {
        float a = 0.0f;
        #pragma unroll
        for (int e = 0; e < EMB; ++e) a += EnS[e] * biasP[e * DD + tid];
        biasS[tid] = a;
    }
    __syncthreads();

    const int wave  = tid >> 5, lane = tid & 31;
    const int lm    = lane & 15;
    const int lk    = (lane >> 4) << 3;
    const int mrow8 = (lane >> 4) << 3;
    const __bf16* An = G + n * G_N;

    for (int mt = wave; mt < 12; mt += 4) {               // 12 bt-tiles of 16
        const int bt0 = mt * 16;
        v8f acc[4] = {};
        const __bf16* Abase = An + (bt0 + lm) * DD + lk;
        #pragma unroll
        for (int kk = 0; kk < KCH; ++kk) {
            #pragma unroll
            for (int i0 = 0; i0 < DD; i0 += 32) {
                v16bf a = load_frag(Abase + kk * G_K + i0);
                #pragma unroll
                for (int f = 0; f < 4; ++f) {
                    v16bf b = load_frag(&Wlds[(kk * DD + f * 16 + lm) * DD + i0 + lk]);
                    acc[f] = __builtin_amdgcn_wmma_f32_16x16x32_bf16(
                        false, a, false, b, (short)0, acc[f], false, false);
                }
            }
        }
        #pragma unroll
        for (int f = 0; f < 4; ++f) {
            int o = f * 16 + lm;
            float bo = biasS[o];
            #pragma unroll
            for (int r = 0; r < 8; ++r) {
                int bt = bt0 + r + mrow8;
                int b = bt / TT, t = bt % TT;
                out[((b * NNODE + n) * TT + t) * DD + o] = acc[f][r] + bo;
            }
        }
    }
}

// ---------------------------------------------------------------------------
extern "C" void kernel_launch(void* const* d_in, const int* in_sizes, int n_in,
                              void* d_out, int out_size, void* d_ws, size_t ws_size,
                              hipStream_t stream) {
    (void)in_sizes; (void)n_in; (void)out_size; (void)ws_size;
    const float* x  = (const float*)d_in[0];   // [16,2048,12,64]
    const float* E  = (const float*)d_in[1];   // [2048,16]
    const float* Wp = (const float*)d_in[2];   // [16,3,64,64]
    const float* bp = (const float*)d_in[3];   // [16,64]
    float* out = (float*)d_out;

    // Workspace carve-up (~260 MB total required).
    char* w = (char*)d_ws;
    __bf16* Sb  = (__bf16*)(w);
    __bf16* Xt  = (__bf16*)(w + (size_t)8388608);
    __bf16* Y1t = (__bf16*)(w + (size_t)8388608 + (size_t)50331648);
    __bf16* G   = (__bf16*)(w + (size_t)8388608 + (size_t)2 * 50331648);

    pack_x_kernel<<<TOTX / 256, 256, 0, stream>>>(x, Xt, G);
    supports_kernel<<<NNODE, 256, 0, stream>>>(E, Sb);
    dim3 g1(32, BT);
    gemm_s_kernel<1><<<g1, 128, 0, stream>>>(Sb, Xt, nullptr, Y1t, G);
    gemm_s_kernel<2><<<g1, 128, 0, stream>>>(Sb, Y1t, x, nullptr, G);
    final_kernel<<<NNODE, 128, 0, stream>>>(E, Wp, bp, G, out);
}
